// RNN_1743756722944
// MI455X (gfx1250) — matrix operand order repaired and checked by reading, compiled, not run on hardware
//
#include <hip/hip_runtime.h>
#include <hip/hip_bf16.h>

// ---------------------------------------------------------------------------
// Problem constants (match reference): B=64, T=1024, I=128, H=256, O=10, L=10
// ---------------------------------------------------------------------------
#define RB   64
#define RT   1024
#define RI   128
#define RH   256
#define RO   10
#define RL   10
#define RBT  (RB * RT)          // 65536 rows in the big GEMMs

typedef __attribute__((ext_vector_type(16))) __bf16 v16bf;
typedef __attribute__((ext_vector_type(8)))  __bf16 v8bf;
typedef __attribute__((ext_vector_type(8)))  float  v8f;
typedef __attribute__((ext_vector_type(4)))  int    v4i;

// ---------------------------------------------------------------------------
// gfx1250 feature probes (compile-safe: fall back when builtins are absent)
// ---------------------------------------------------------------------------
#if defined(__HIP_DEVICE_COMPILE__) &&                                         \
    __has_builtin(__builtin_amdgcn_global_load_async_to_lds_b128) &&           \
    __has_builtin(__builtin_amdgcn_s_wait_asynccnt)
#define USE_ASYNC 1
#else
#define USE_ASYNC 0
#endif

// 16B global -> LDS copy chunk. Async path: per-lane DMA into LDS, no VGPR
// staging, tracked by ASYNCcnt. Fallback: load to regs + ds_store.
// Builtin signature (from hipcc diagnostic): param0 = v4i* in AS(1) [global],
// param1 = v4i* in AS(3) [LDS], then imm offset, imm cpol.
__device__ __forceinline__ void cp16(__bf16* dst_lds, const __bf16* src_glb) {
#if USE_ASYNC
    __builtin_amdgcn_global_load_async_to_lds_b128(
        (__attribute__((address_space(1))) v4i*)src_glb,
        (__attribute__((address_space(3))) v4i*)dst_lds, 0, 0);
#else
    *(uint4*)dst_lds = *(const uint4*)src_glb;
#endif
}

__device__ __forceinline__ void cp_wait() {
#if USE_ASYNC
    __builtin_amdgcn_s_wait_asynccnt(0);
#endif
}

// gfx1250 hardware tanh (v_tanh_f32) when available.
__device__ __forceinline__ float fast_tanh(float x) {
#if __has_builtin(__builtin_amdgcn_tanhf)
    return __builtin_amdgcn_tanhf(x);
#elif __has_builtin(__builtin_amdgcn_tanh_f32)
    return __builtin_amdgcn_tanh_f32(x);
#else
    return tanhf(x);
#endif
}

// ---------------------------------------------------------------------------
// WMMA fragment helpers (CDNA5 16x16x32 bf16, wave32; contiguous k order, not ISA table order)
// A (16x32, row-major, leading-dim ldk):
//   lane = half*16+m ; elements 0..15 -> K = half*16 + 0..15  (contiguous)
// B (32x16 = K x N) taken from a row-major [N,K] weight (computes X * W^T):
//   lane = half*16+n ; elements 0..15 -> K = half*16 + 0..15  (contiguous)
// ---------------------------------------------------------------------------
__device__ __forceinline__ v16bf load_a_frag(const __bf16* row_base, int ldk, int lane) {
    const int m = lane & 15, half = lane >> 4;
    const __bf16* p = row_base + (size_t)m * ldk;
    v8bf lo = *(const v8bf*)(p + half * 16);
    v8bf hi = *(const v8bf*)(p + half * 16 + 8);
    return __builtin_shufflevector(lo, hi, 0,1,2,3,4,5,6,7,8,9,10,11,12,13,14,15);
}

__device__ __forceinline__ v16bf load_b_frag(const __bf16* row_base, int ldk, int lane) {
    const int n = lane & 15, half = lane >> 4;
    const __bf16* p = row_base + (size_t)n * ldk + half * 16;
    return *(const v16bf*)p;
}

__device__ __forceinline__ v8f wmma_bf16(v16bf a, v16bf b, v8f c) {
    return __builtin_amdgcn_wmma_f32_16x16x32_bf16(false, a, false, b, (short)0, c,
                                                   false, false);
}

// ---------------------------------------------------------------------------
// Elementwise converts (run once per launch; deterministic)
// ---------------------------------------------------------------------------
__global__ void cvt_f32_to_bf16(const float* __restrict__ in,
                                __bf16* __restrict__ out, int n) {
    int i = blockIdx.x * blockDim.x + threadIdx.x;
    if (i < n) out[i] = (__bf16)in[i];
}

// x [B,T,128] f32  ->  act rows [B*T, 256] bf16 (cols 0..127 valid)
__global__ void cvt_x_to_act(const float* __restrict__ x, __bf16* __restrict__ act) {
    int i = blockIdx.x * blockDim.x + threadIdx.x;   // over B*T*128
    int row = i >> 7, c = i & 127;
    act[(size_t)row * RH + c] = (__bf16)x[i];
}

// ---------------------------------------------------------------------------
// Projection GEMM:  pre[M,256] = act[M,K](bf16) * W[256,K]^T + (bih + bhh)
// Block: 256 thr = 8 waves on a 128x64 tile (wave grid 4x2, 32x32 per wave).
// Double-buffered LDS, K chunk = 32; next chunk streams in (async-to-LDS on
// gfx1250) while WMMAs consume the current one. One barrier per chunk.
// ---------------------------------------------------------------------------
#define LDA 40   // padded LDS leading dim (bf16 elems); 80B rows, 16B aligned
__global__ __launch_bounds__(256) void proj_gemm(
        const __bf16* __restrict__ act, const __bf16* __restrict__ W,
        const float* __restrict__ bih, const float* __restrict__ bhh,
        float* __restrict__ pre, int K) {
    __shared__ __align__(16) __bf16 As[2][128 * LDA];
    __shared__ __align__(16) __bf16 Bs[2][64 * LDA];

    const int tid  = threadIdx.x;
    const int lane = tid & 31;
    const int wave = tid >> 5;
    const int wm   = wave & 3;      // 4 wave-rows  (32 M each)
    const int wn   = wave >> 2;     // 2 wave-cols  (32 N each)
    const int m0   = blockIdx.x * 128;
    const int n0   = blockIdx.y * 64;

    // per-thread copy assignments (A: 2 chunks, B: 1 chunk, 16B each)
    const int ar0 = tid >> 1, aq0 = (tid & 1) * 2;       // rows 0..127, quarters {0,2}
    const int br  = tid >> 2, bq  = tid & 3;             // rows 0..63,  quarter 0..3

    v8f acc[2][2] = {};
    const int nk = K / 32;

    // prologue: stage chunk 0 into buffer 0
    cp16(&As[0][ar0 * LDA + aq0 * 8], act + (size_t)(m0 + ar0) * RH + aq0 * 8);
    cp16(&As[0][ar0 * LDA + (aq0 + 1) * 8], act + (size_t)(m0 + ar0) * RH + (aq0 + 1) * 8);
    cp16(&Bs[0][br * LDA + bq * 8], W + (size_t)(n0 + br) * K + bq * 8);
    cp_wait();
    __syncthreads();

    for (int ki = 0; ki < nk; ++ki) {
        const int cur = ki & 1, nxt = cur ^ 1;
        if (ki + 1 < nk) {   // stream next chunk while computing this one
            const int k0 = (ki + 1) * 32;
            cp16(&As[nxt][ar0 * LDA + aq0 * 8],
                 act + (size_t)(m0 + ar0) * RH + k0 + aq0 * 8);
            cp16(&As[nxt][ar0 * LDA + (aq0 + 1) * 8],
                 act + (size_t)(m0 + ar0) * RH + k0 + (aq0 + 1) * 8);
            cp16(&Bs[nxt][br * LDA + bq * 8],
                 W + (size_t)(n0 + br) * K + k0 + bq * 8);
        }
        #pragma unroll
        for (int tm = 0; tm < 2; ++tm) {
            v16bf a = load_a_frag(&As[cur][(wm * 32 + tm * 16) * LDA], LDA, lane);
            #pragma unroll
            for (int tn = 0; tn < 2; ++tn) {
                v16bf b = load_b_frag(&Bs[cur][(wn * 32 + tn * 16) * LDA], LDA, lane);
                acc[tm][tn] = wmma_bf16(a, b, acc[tm][tn]);
            }
        }
        cp_wait();
        __syncthreads();
    }

    // Epilogue: fused bias add, f32 store. C/D layout: m = r + 8*half, n = lane&15.
    const int nl = lane & 15, half = lane >> 4;
    #pragma unroll
    for (int tn = 0; tn < 2; ++tn) {
        int n = n0 + wn * 32 + tn * 16 + nl;
        float bsum = bih[n] + bhh[n];
        #pragma unroll
        for (int tm = 0; tm < 2; ++tm) {
            #pragma unroll
            for (int r = 0; r < 8; ++r) {
                int m = m0 + wm * 32 + tm * 16 + r + 8 * half;
                pre[(size_t)m * RH + n] = acc[tm][tn][r] + bsum;
            }
        }
    }
}

// ---------------------------------------------------------------------------
// Recurrent scan: h = tanh(pre_t + h * Whh^T), 1024 sequential steps.
// Batch split: 4 workgroups x 16 batch rows (rows independent -> no
// cross-WGP sync). 8 waves, each owns a 16x32 N-slice (2 WMMA tiles).
//  - Whh B-fragments are loop-invariant -> hoisted into VGPRs (512B/lane);
//    the steady-state loop touches global memory only for pre / actOut.
//  - h double-buffered in LDS (bf16, padded stride) -> single barrier/step.
//  - pre_t loaded before the WMMA phase so HBM latency hides behind WMMAs.
//  - hardware v_tanh_f32 when available.
// ---------------------------------------------------------------------------
#define LDH 264  // 528B rows: 16B aligned, distinct-bank A-frag reads
__global__ __launch_bounds__(256) void rnn_scan(
        const float* __restrict__ pre, const __bf16* __restrict__ Whh,
        const float* __restrict__ h0, __bf16* __restrict__ actOut,
        float* __restrict__ hFinal) {
    __shared__ __align__(16) __bf16 Hs[2][16 * LDH];

    const int tid  = threadIdx.x;
    const int lane = tid & 31;
    const int wn   = tid >> 5;            // wave id = N-slice (32 cols each)
    const int b0   = blockIdx.x * 16;     // batch slice
    const int nl   = lane & 15, half = lane >> 4;

    // init h (t=0 reads buffer 0)
    for (int i = tid; i < 16 * RH; i += 256) {
        int r = i >> 8, c = i & 255;
        Hs[0][r * LDH + c] = (__bf16)h0[(size_t)(b0 + r) * RH + c];
    }

    // hoist this wave's Whh fragments into registers (invariant over t)
    v16bf breg[8][2];
    #pragma unroll
    for (int kk = 0; kk < 8; ++kk)
        #pragma unroll
        for (int tn = 0; tn < 2; ++tn)
            breg[kk][tn] = load_b_frag(
                Whh + (size_t)(wn * 32 + tn * 16) * RH + kk * 32, RH, lane);

    __syncthreads();

    for (int t = 0; t < RT; ++t) {
        const int cur = t & 1, nxt = cur ^ 1;

        // independent global loads first: latency hides behind the WMMA chain
        float pv[2][8];
        #pragma unroll
        for (int tn = 0; tn < 2; ++tn)
            #pragma unroll
            for (int r = 0; r < 8; ++r)
                pv[tn][r] = pre[((size_t)(b0 + r + 8 * half) * RT + t) * RH
                                + (wn * 32 + tn * 16 + nl)];

        v8f acc[2] = {};
        #pragma unroll
        for (int kk = 0; kk < 8; ++kk) {
            v16bf a = load_a_frag(&Hs[cur][kk * 32], LDH, lane);
            acc[0] = wmma_bf16(a, breg[kk][0], acc[0]);
            acc[1] = wmma_bf16(a, breg[kk][1], acc[1]);
        }

        #pragma unroll
        for (int tn = 0; tn < 2; ++tn) {
            int n = wn * 32 + tn * 16 + nl;
            #pragma unroll
            for (int r = 0; r < 8; ++r) {
                int m = r + 8 * half;             // row within batch slice
                int b = b0 + m;
                float hnew = fast_tanh(acc[tn][r] + pv[tn][r]);
                __bf16 hb = (__bf16)hnew;
                Hs[nxt][m * LDH + n] = hb;        // next step's A operand
                actOut[((size_t)b * RT + t) * RH + n] = hb;  // next layer input
                if (t == RT - 1) hFinal[(size_t)b * RH + n] = hnew;
            }
        }
        __syncthreads();   // Hs[nxt] complete before next step reads it
    }
}

// ---------------------------------------------------------------------------
// Tiny final FC: y[64,10] = h_last[64,256] * fc_w^T + fc_b   (327.7 KFLOP)
// ---------------------------------------------------------------------------
__global__ void fc_kernel(const float* __restrict__ hlast,
                          const float* __restrict__ fcw,
                          const float* __restrict__ fcb,
                          float* __restrict__ y) {
    int i = blockIdx.x * blockDim.x + threadIdx.x;
    if (i >= RB * RO) return;
    int b = i / RO, o = i % RO;
    float s = fcb[o];
    #pragma unroll 8
    for (int k = 0; k < RH; ++k) s += hlast[(size_t)b * RH + k] * fcw[o * RH + k];
    y[b * RO + o] = s;
}

// ---------------------------------------------------------------------------
// Host orchestration
// ---------------------------------------------------------------------------
extern "C" void kernel_launch(void* const* d_in, const int* in_sizes, int n_in,
                              void* d_out, int out_size, void* d_ws, size_t ws_size,
                              hipStream_t stream) {
    const float* x     = (const float*)d_in[0];
    const float* h0    = (const float*)d_in[1];
    const float* w_ih0 = (const float*)d_in[2];
    const float* w_ih  = (const float*)d_in[3];   // [9,256,256]
    const float* w_hh  = (const float*)d_in[4];   // [10,256,256]
    const float* b_ih  = (const float*)d_in[5];   // [10,256]
    const float* b_hh  = (const float*)d_in[6];   // [10,256]
    const float* fc_w  = (const float*)d_in[7];   // [10,256]
    const float* fc_b  = (const float*)d_in[8];   // [10]
    float* out = (float*)d_out;                   // y[640] then h[10*64*256]

    // Workspace layout
    char*   ws   = (char*)d_ws;
    float*  pre  = (float*)ws;                                   // 64 MB
    __bf16* act  = (__bf16*)(ws + (size_t)RBT * RH * 4);         // 32 MB
    __bf16* wihb = (__bf16*)(ws + (size_t)RBT * RH * 4
                                + (size_t)RBT * RH * 2);         // 1.2 MB
    __bf16* whhb = wihb + (RH * RI + 9 * RH * RH);               // 1.3 MB

    const int n_wih0 = RH * RI;            // 32768
    const int n_wih  = 9 * RH * RH;        // 589824
    const int n_whh  = RL * RH * RH;       // 655360

    cvt_f32_to_bf16<<<(n_wih0 + 255) / 256, 256, 0, stream>>>(w_ih0, wihb, n_wih0);
    cvt_f32_to_bf16<<<(n_wih  + 255) / 256, 256, 0, stream>>>(w_ih, wihb + n_wih0, n_wih);
    cvt_f32_to_bf16<<<(n_whh  + 255) / 256, 256, 0, stream>>>(w_hh, whhb, n_whh);
    cvt_x_to_act<<<(RBT * RI) / 256, 256, 0, stream>>>(x, act);

    float* h_out = out + RB * RO;   // d_out h-section base

    for (int l = 0; l < RL; ++l) {
        const int K = (l == 0) ? RI : RH;
        const __bf16* W = (l == 0) ? wihb : wihb + n_wih0 + (size_t)(l - 1) * RH * RH;

        dim3 pg(RBT / 128, RH / 64);
        proj_gemm<<<pg, 256, 0, stream>>>(act, W, b_ih + l * RH, b_hh + l * RH,
                                          pre, K);
        rnn_scan<<<RB / 16, 256, 0, stream>>>(pre, whhb + (size_t)l * RH * RH,
                                              h0 + (size_t)l * RB * RH,
                                              act, h_out + (size_t)l * RB * RH);
    }

    fc_kernel<<<(RB * RO + 255) / 256, 256, 0, stream>>>(
        h_out + (size_t)(RL - 1) * RB * RH, fc_w, fc_b, out);
}
